// set_transformer_model_82506321756813
// MI455X (gfx1250) — compile-verified, matrix operand order checked by reading, not run on hardware
//
#include <hip/hip_runtime.h>
#include <hip/hip_bf16.h>
#include <math.h>

// ---------------------------------------------------------------------------
// Set Transformer forward on gfx1250 (MI455X), WMMA f16->f32 matrix cores.
// ---------------------------------------------------------------------------

typedef __attribute__((ext_vector_type(16))) _Float16 v16h;
typedef __attribute__((ext_vector_type(8)))  _Float16 v8h;
typedef __attribute__((ext_vector_type(8)))  float    v8f;

#define NKEYS 2048
#define EMB   768
#define HID   512
#define HEADS 8
#define HD    64
#define NOUT  7
#define PMA_M 32   // PMA seed rows padded to one 32-row GEMM tile

__device__ __forceinline__ v8f wmma32f16(v16h a, v16h b, v8f c) {
  // D = A(16x32 f16) * B(32x16 f16) + C(16x16 f32)
  return __builtin_amdgcn_wmma_f32_16x16x32_f16(
      /*neg_a=*/false, a, /*neg_b=*/false, b,
      /*c_mod=*/(short)0, c, /*reuse_a=*/false, /*reuse_b=*/false);
}

// Load a 16-half A/B fragment for v_wmma_f32_16x16x32_f16.
// Per ISA layout: lanes 0-15 hold K {kbase+0..7, kbase+16..23} of row `row`,
// lanes 16-31 hold K {kbase+8..15, kbase+24..31}. Two 16B contiguous loads.
__device__ __forceinline__ v16h load_frag(const _Float16* base, int ld,
                                          int row, int kbase) {
  const int lane = threadIdx.x & 31;
  const _Float16* r = base + (size_t)row * ld + kbase + ((lane & 16) ? 8 : 0);
  union { v16h v; v8h h[2]; } u;
  u.h[0] = *reinterpret_cast<const v8h*>(r);
  u.h[1] = *reinterpret_cast<const v8h*>(r + 16);
  return u.v;
}

// ---------------------------------------------------------------------------
// f32 -> f16 elementwise conversion (weights)
// ---------------------------------------------------------------------------
__global__ void convert_f16_kernel(const float* __restrict__ src,
                                   _Float16* __restrict__ dst, int n) {
  int i = blockIdx.x * blockDim.x + threadIdx.x;
  if (i < n) dst[i] = (_Float16)src[i];
}

// ---------------------------------------------------------------------------
// Embedding gather: emb[key][0..767] f16
// ---------------------------------------------------------------------------
__global__ void embed_kernel(const int* __restrict__ is_cat,
                             const int* __restrict__ cat_idx,
                             const int* __restrict__ reg_idx,
                             const float* __restrict__ reg_values,
                             const float* __restrict__ cat_table,
                             const float* __restrict__ reg_table,
                             const float* __restrict__ regW,
                             const float* __restrict__ regb,
                             _Float16* __restrict__ emb) {
  const int key = blockIdx.x;
  const int ci = cat_idx[key];
  const int ri = reg_idx[key];
  const bool cat = is_cat[key] != 0;
  const float rv = reg_values[key];
  for (int e = threadIdx.x; e < EMB; e += blockDim.x) {
    float v;
    if (cat) {
      v = cat_table[(size_t)ci * EMB + e];
    } else if (e < 512) {
      v = reg_table[(size_t)ri * 512 + e];
    } else {
      v = regW[(size_t)ri * 256 + (e - 512)] * rv +
          regb[(size_t)ri * 256 + (e - 512)];
    }
    emb[(size_t)key * EMB + e] = (_Float16)v;
  }
}

// ---------------------------------------------------------------------------
// WMMA GEMM: C[M,N] = A[M,K](f16) * W[N,K](f16)^T + bias
//   one wave computes a 32x64 tile: 2 A fragments x 4 B fragments per k-step
//   -> 8 independent WMMAs per 6 fragment loads (latency hiding + less B
//   re-fetch).  M must be a multiple of 32.
//   epilogue options: f32 out, f16 out, transposed-f16 out, relu-residual
// ---------------------------------------------------------------------------
__global__ __launch_bounds__(32)
void gemm32_kernel(const _Float16* __restrict__ A,
                   const _Float16* __restrict__ W,
                   const float* __restrict__ bias,
                   int M, int N, int K,
                   float* __restrict__ Cf32,
                   _Float16* __restrict__ Cf16,
                   _Float16* __restrict__ Ct16,
                   const float* __restrict__ Rres) {
  const int lane = threadIdx.x & 31;
  const int mt = blockIdx.x * 32;
  const int nt = blockIdx.y * 64;

  v8f acc0[4] = {};
  v8f acc1[4] = {};
  const int arow0 = mt + (lane & 15);
  const int arow1 = mt + 16 + (lane & 15);
  for (int k = 0; k < K; k += 32) {
    v16h a0 = load_frag(A, K, arow0, k);
    v16h a1 = load_frag(A, K, arow1, k);
#pragma unroll
    for (int j = 0; j < 4; ++j) {
      v16h b = load_frag(W, K, nt + j * 16 + (lane & 15), k);
      acc0[j] = wmma32f16(a0, b, acc0[j]);
      acc1[j] = wmma32f16(a1, b, acc1[j]);
    }
  }

  // C/D layout: lanes 0-15: N=lane, M=v; lanes 16-31: N=lane-16, M=v+8
  const int ncol = nt + (lane & 15);
#pragma unroll
  for (int h = 0; h < 2; ++h) {
    const int mrow = mt + h * 16 + ((lane >> 4) << 3);
#pragma unroll
    for (int j = 0; j < 4; ++j) {
      const int n = ncol + j * 16;
      const float bv = bias ? bias[n] : 0.0f;
      const v8f& av = h ? acc1[j] : acc0[j];
#pragma unroll
      for (int v = 0; v < 8; ++v) {
        const int m = mrow + v;
        float val = av[v] + bv;
        if (Rres) val = Rres[(size_t)m * N + n] + fmaxf(val, 0.0f);
        if (Cf32) Cf32[(size_t)m * N + n] = val;
        if (Cf16) Cf16[(size_t)m * N + n] = (_Float16)val;
        if (Ct16) Ct16[(size_t)n * M + m] = (_Float16)val;
      }
    }
  }
}

// ---------------------------------------------------------------------------
// Flash attention per (16-query tile, head), online softmax, residual add.
//   Q16/K16: [rows, HID] f16 projections; Vt: [HID, NK] f16 (transposed V)
//   Out: O = Q32 + softmax(Q K^T * scale) V   (f32 + f16 copies)
// ---------------------------------------------------------------------------
__global__ __launch_bounds__(32)
void attn_kernel(const _Float16* __restrict__ Q16,
                 const float* __restrict__ Q32,
                 const _Float16* __restrict__ K16,
                 const _Float16* __restrict__ Vt,
                 float* __restrict__ O32,
                 _Float16* __restrict__ O16,
                 int NK, float scale) {
  __shared__ __align__(16) _Float16 Pl[16 * 32];  // P tile (q x 32 keys)
  const int lane = threadIdx.x & 31;
  const int qt = blockIdx.x * 16;
  const int dbase = blockIdx.y * HD;
  const int rsel = (lane >> 4) << 3;  // +8 rows for upper half-wave

  // Q fragments (A operand) over the 64-wide head slice, loaded once.
  const v16h qf0 = load_frag(Q16 + dbase, HID, qt + (lane & 15), 0);
  const v16h qf1 = load_frag(Q16 + dbase, HID, qt + (lane & 15), 32);

  float mrow[8], lrow[8];
  v8f oacc[4] = {};
#pragma unroll
  for (int v = 0; v < 8; ++v) { mrow[v] = -1e30f; lrow[v] = 0.0f; }

  for (int kb = 0; kb < NK; kb += 32) {
    // S = Q K^T for 32 keys -> two 16x16 f32 tiles
    v8f s0 = {}, s1 = {};
    {
      v16h b0 = load_frag(K16 + dbase, HID, kb + (lane & 15), 0);
      v16h b1 = load_frag(K16 + dbase, HID, kb + (lane & 15), 32);
      s0 = wmma32f16(qf0, b0, s0);
      s0 = wmma32f16(qf1, b1, s0);
      b0 = load_frag(K16 + dbase, HID, kb + 16 + (lane & 15), 0);
      b1 = load_frag(K16 + dbase, HID, kb + 16 + (lane & 15), 32);
      s1 = wmma32f16(qf0, b0, s1);
      s1 = wmma32f16(qf1, b1, s1);
    }

    // Online softmax: per-row max/sum, rescale accumulators, stage P in LDS.
    float ps[8];
#pragma unroll
    for (int v = 0; v < 8; ++v) {
      const float a0 = s0[v] * scale;
      const float a1 = s1[v] * scale;
      float mx = fmaxf(a0, a1);
#pragma unroll
      for (int off = 1; off < 16; off <<= 1)
        mx = fmaxf(mx, __shfl_xor(mx, off, 32));
      const float nm = fmaxf(mrow[v], mx);
      const float corr = __expf(mrow[v] - nm);
      lrow[v] *= corr;
#pragma unroll
      for (int g = 0; g < 4; ++g) oacc[g][v] *= corr;
      mrow[v] = nm;
      const float p0 = __expf(a0 - nm);
      const float p1 = __expf(a1 - nm);
      Pl[(rsel + v) * 32 + (lane & 15)] = (_Float16)p0;
      Pl[(rsel + v) * 32 + 16 + (lane & 15)] = (_Float16)p1;
      ps[v] = p0 + p1;
    }
#pragma unroll
    for (int v = 0; v < 8; ++v) {
      float a = ps[v];
#pragma unroll
      for (int off = 1; off < 16; off <<= 1) a += __shfl_xor(a, off, 32);
      lrow[v] += a;
    }
    __syncthreads();
    const v16h pf = load_frag((const _Float16*)Pl, 32, lane & 15, 0);
    __syncthreads();

    // O += P (16x32) * V (32x64): B operand rows are contiguous in Vt.
#pragma unroll
    for (int g = 0; g < 4; ++g) {
      v16h vf = load_frag(Vt, NK, dbase + g * 16 + (lane & 15), kb);
      oacc[g] = wmma32f16(pf, vf, oacc[g]);
    }
  }

  // Epilogue: O = Q + oacc / l   (residual from f32 Q projection)
  const int n0 = lane & 15;
#pragma unroll
  for (int g = 0; g < 4; ++g) {
    const int col = dbase + g * 16 + n0;
#pragma unroll
    for (int v = 0; v < 8; ++v) {
      const int row = qt + rsel + v;
      const float val =
          Q32[(size_t)row * HID + col] + oacc[g][v] / lrow[v];
      O32[(size_t)row * HID + col] = val;
      O16[(size_t)row * HID + col] = (_Float16)val;
    }
  }
}

// ---------------------------------------------------------------------------
// Zero-pad PMA seeds S [7,512] f32 -> [PMA_M,512] f16
// ---------------------------------------------------------------------------
__global__ void pad_seed_kernel(const float* __restrict__ S,
                                _Float16* __restrict__ Spad) {
  int i = blockIdx.x * blockDim.x + threadIdx.x;
  if (i < PMA_M * HID) {
    int r = i / HID;
    Spad[i] = (r < NOUT) ? (_Float16)S[i] : (_Float16)0.0f;
  }
}

// ---------------------------------------------------------------------------
// Final Linear(512,1): out[i] = dot(P[i,:], fc_w) + fc_b
// ---------------------------------------------------------------------------
__global__ __launch_bounds__(32)
void fc_kernel(const float* __restrict__ P, const float* __restrict__ fcw,
               const float* __restrict__ fcb, float* __restrict__ out) {
  const int row = blockIdx.x;
  const int lane = threadIdx.x & 31;
  float s = 0.0f;
  for (int c = lane; c < HID; c += 32) s += P[row * HID + c] * fcw[c];
#pragma unroll
  for (int off = 16; off; off >>= 1) s += __shfl_xor(s, off, 32);
  if (lane == 0) out[row] = s + fcb[0];
}

// ---------------------------------------------------------------------------
// Host launcher
// ---------------------------------------------------------------------------
extern "C" void kernel_launch(void* const* d_in, const int* in_sizes, int n_in,
                              void* d_out, int out_size, void* d_ws,
                              size_t ws_size, hipStream_t stream) {
  (void)in_sizes; (void)n_in; (void)out_size; (void)ws_size;
  const int*   is_cat    = (const int*)d_in[0];
  const int*   cat_idx   = (const int*)d_in[1];
  const int*   reg_idx   = (const int*)d_in[2];
  const float* reg_vals  = (const float*)d_in[3];
  const float* cat_table = (const float*)d_in[4];
  const float* reg_table = (const float*)d_in[5];
  const float* reg_W     = (const float*)d_in[6];
  const float* reg_b     = (const float*)d_in[7];
  const float* q0_w = (const float*)d_in[8];  const float* q0_b = (const float*)d_in[9];
  const float* k0_w = (const float*)d_in[10]; const float* k0_b = (const float*)d_in[11];
  const float* v0_w = (const float*)d_in[12]; const float* v0_b = (const float*)d_in[13];
  const float* o0_w = (const float*)d_in[14]; const float* o0_b = (const float*)d_in[15];
  const float* qs_w = (const float*)d_in[16]; const float* qs_b = (const float*)d_in[17];
  const float* ks_w = (const float*)d_in[18]; const float* ks_b = (const float*)d_in[19];
  const float* vs_w = (const float*)d_in[20]; const float* vs_b = (const float*)d_in[21];
  const float* os_w = (const float*)d_in[22]; const float* os_b = (const float*)d_in[23];
  const float* Sseed = (const float*)d_in[24];
  const float* pq_w = (const float*)d_in[25]; const float* pq_b = (const float*)d_in[26];
  const float* pk_w = (const float*)d_in[27]; const float* pk_b = (const float*)d_in[28];
  const float* pv_w = (const float*)d_in[29]; const float* pv_b = (const float*)d_in[30];
  const float* po_w = (const float*)d_in[31]; const float* po_b = (const float*)d_in[32];
  const float* fc_w = (const float*)d_in[33]; const float* fc_b = (const float*)d_in[34];
  float* out = (float*)d_out;

  // ---- workspace layout (bump allocator, 256B aligned) ----
  char* ws = (char*)d_ws;
  size_t off = 0;
  auto alloc = [&](size_t bytes) -> void* {
    void* p = ws + off;
    off += (bytes + 255) & ~(size_t)255;
    return p;
  };
  _Float16* wq0  = (_Float16*)alloc((size_t)HID * EMB * 2);
  _Float16* wk0  = (_Float16*)alloc((size_t)HID * EMB * 2);
  _Float16* wv0  = (_Float16*)alloc((size_t)HID * EMB * 2);
  _Float16* wo0  = (_Float16*)alloc((size_t)HID * HID * 2);
  _Float16* wqs  = (_Float16*)alloc((size_t)3 * HID * HID * 2);
  _Float16* wks  = (_Float16*)alloc((size_t)3 * HID * HID * 2);
  _Float16* wvs  = (_Float16*)alloc((size_t)3 * HID * HID * 2);
  _Float16* wos  = (_Float16*)alloc((size_t)3 * HID * HID * 2);
  _Float16* wpq  = (_Float16*)alloc((size_t)HID * HID * 2);
  _Float16* wpk  = (_Float16*)alloc((size_t)HID * HID * 2);
  _Float16* wpv  = (_Float16*)alloc((size_t)HID * HID * 2);
  _Float16* wpo  = (_Float16*)alloc((size_t)HID * HID * 2);
  _Float16* emb16 = (_Float16*)alloc((size_t)NKEYS * EMB * 2);
  float*    Qp32  = (float*)alloc((size_t)NKEYS * HID * 4);
  _Float16* Qp16  = (_Float16*)alloc((size_t)NKEYS * HID * 2);
  _Float16* Kp16  = (_Float16*)alloc((size_t)NKEYS * HID * 2);
  _Float16* Vt16  = (_Float16*)alloc((size_t)HID * NKEYS * 2);
  float*    O32   = (float*)alloc((size_t)NKEYS * HID * 4);
  _Float16* O16   = (_Float16*)alloc((size_t)NKEYS * HID * 2);
  _Float16* X16   = (_Float16*)alloc((size_t)NKEYS * HID * 2);
  _Float16* Spad16 = (_Float16*)alloc((size_t)PMA_M * HID * 2);
  float*    Qs32   = (float*)alloc((size_t)PMA_M * HID * 4);
  _Float16* Qs16   = (_Float16*)alloc((size_t)PMA_M * HID * 2);
  float*    P32    = (float*)alloc((size_t)PMA_M * HID * 4);
  _Float16* P16    = (_Float16*)alloc((size_t)PMA_M * HID * 2);
  float*    F32    = (float*)alloc((size_t)PMA_M * HID * 4);

  const float scale = 0.044194173824159216f;  // 1/sqrt(512)

  auto conv = [&](const float* src, _Float16* dst, int n) {
    convert_f16_kernel<<<(n + 255) / 256, 256, 0, stream>>>(src, dst, n);
  };
  auto gemm = [&](const _Float16* A, const _Float16* W, const float* b, int M,
                  int N, int K, float* c32, _Float16* c16, _Float16* ct16,
                  const float* rres) {
    gemm32_kernel<<<dim3(M / 32, N / 64), 32, 0, stream>>>(
        A, W, b, M, N, K, c32, c16, ct16, rres);
  };
  auto attn = [&](const _Float16* q16, const float* q32, const _Float16* k16,
                  const _Float16* vt, float* o32, _Float16* o16, int Mq) {
    attn_kernel<<<dim3(Mq / 16, HEADS), 32, 0, stream>>>(q16, q32, k16, vt,
                                                         o32, o16, NKEYS,
                                                         scale);
  };

  // ---- weight conversion (once per call; cheap, L2 resident) ----
  conv(q0_w, wq0, HID * EMB);
  conv(k0_w, wk0, HID * EMB);
  conv(v0_w, wv0, HID * EMB);
  conv(o0_w, wo0, HID * HID);
  conv(qs_w, wqs, 3 * HID * HID);
  conv(ks_w, wks, 3 * HID * HID);
  conv(vs_w, wvs, 3 * HID * HID);
  conv(os_w, wos, 3 * HID * HID);
  conv(pq_w, wpq, HID * HID);
  conv(pk_w, wpk, HID * HID);
  conv(pv_w, wpv, HID * HID);
  conv(po_w, wpo, HID * HID);

  // ---- embedding gather ----
  embed_kernel<<<NKEYS, 256, 0, stream>>>(is_cat, cat_idx, reg_idx, reg_vals,
                                          cat_table, reg_table, reg_W, reg_b,
                                          emb16);

  // ---- encoder layer 0 (input dim 768) ----
  gemm(emb16, wq0, q0_b, NKEYS, HID, EMB, Qp32, Qp16, nullptr, nullptr);
  gemm(emb16, wk0, k0_b, NKEYS, HID, EMB, nullptr, Kp16, nullptr, nullptr);
  gemm(emb16, wv0, v0_b, NKEYS, HID, EMB, nullptr, nullptr, Vt16, nullptr);
  attn(Qp16, Qp32, Kp16, Vt16, O32, O16, NKEYS);
  gemm(O16, wo0, o0_b, NKEYS, HID, HID, nullptr, X16, nullptr, O32);

  // ---- encoder layers 1..3 (input dim 512) ----
  for (int i = 0; i < 3; ++i) {
    const _Float16* wq = wqs + (size_t)i * HID * HID;
    const _Float16* wk = wks + (size_t)i * HID * HID;
    const _Float16* wv = wvs + (size_t)i * HID * HID;
    const _Float16* wo = wos + (size_t)i * HID * HID;
    gemm(X16, wq, qs_b + i * HID, NKEYS, HID, HID, Qp32, Qp16, nullptr, nullptr);
    gemm(X16, wk, ks_b + i * HID, NKEYS, HID, HID, nullptr, Kp16, nullptr, nullptr);
    gemm(X16, wv, vs_b + i * HID, NKEYS, HID, HID, nullptr, nullptr, Vt16, nullptr);
    attn(Qp16, Qp32, Kp16, Vt16, O32, O16, NKEYS);
    gemm(O16, wo, os_b + i * HID, NKEYS, HID, HID, nullptr, X16, nullptr, O32);
  }

  // ---- PMA decoder: MAB(S, X) with 7 seeds padded to one 32-row tile ----
  pad_seed_kernel<<<(PMA_M * HID + 255) / 256, 256, 0, stream>>>(Sseed, Spad16);
  gemm(Spad16, wpq, pq_b, PMA_M, HID, HID, Qs32, Qs16, nullptr, nullptr);
  gemm(X16, wpk, pk_b, NKEYS, HID, HID, nullptr, Kp16, nullptr, nullptr);
  gemm(X16, wpv, pv_b, NKEYS, HID, HID, nullptr, nullptr, Vt16, nullptr);
  attn(Qs16, Qs32, Kp16, Vt16, P32, P16, PMA_M);
  gemm(P16, wpo, po_b, PMA_M, HID, HID, F32, nullptr, nullptr, P32);

  // ---- final Linear(512, 1) over the 7 valid rows ----
  fc_kernel<<<NOUT, 32, 0, stream>>>(F32, fc_w, fc_b, out);
}
